// GraphTransformer_46007689675008
// MI455X (gfx1250) — matrix operand order repaired and checked by reading, not compiled, not run on hardware
//
#include <hip/hip_runtime.h>

// ---------------- problem constants ----------------
#define NT    50000
#define NP    256
#define NE    800000
#define ETOT  850000   // NE + NT self loops
#define EMB   64
#define HEADS 4
#define HE    256      // HEADS*EMB
#define EDIM  8
#define TDIM  16
#define NG    8

typedef __attribute__((ext_vector_type(16))) _Float16 v16h;
typedef __attribute__((ext_vector_type(8)))  float    v8f;

// ---------------- CDNA5 async global->LDS copy (ASYNCcnt-tracked) ----------------
// Copies 16 bytes per active lane from global memory directly into LDS without a
// VGPR round trip (GLOBAL_LOAD_ASYNC_TO_LDS_B128, GV addressing mode).
__device__ __forceinline__ void async_b128(void* lds_dst, const void* gsrc) {
  unsigned l = (unsigned)(uintptr_t)lds_dst; // low 32 bits of generic LDS ptr = LDS offset
  asm volatile("global_load_async_to_lds_b128 %0, %1, off"
               :: "v"(l), "v"((unsigned long long)(uintptr_t)gsrc)
               : "memory");
}
__device__ __forceinline__ void async_wait0() {
  asm volatile("s_wait_asynccnt 0" ::: "memory");
}

// ---------------- WMMA fragment helpers (wave32, 16x16x32 f16) ----------------
// A matrix 16x32 f16: lanes 0-15 -> M=lane, K-halves {0..7,16..23};
//                     lanes 16-31 -> M=lane-16, K-halves {8..15,24..31}
__device__ __forceinline__ v16h frag_a(const _Float16* lds, int lane, int ldm, int kbase) {
  int m = lane & 15;
  int koff = (lane >> 4) << 3; // 0 or 8
  v16h a;
#pragma unroll
  for (int i = 0; i < 16; ++i) {
    int k = ((i >> 3) << 4) + koff + (i & 7);
    a[i] = lds[m * ldm + kbase + k];
  }
  return a;
}
// B matrix 32x16 f16 (K-major [K][N] in LDS): lanes 0-15 hold K={0..7,16..23}, lanes 16-31 K={8..15,24..31}
__device__ __forceinline__ v16h frag_b(const _Float16* lds, int lane, int ldn, int kbase, int nbase) {
  int n = nbase + (lane & 15);
  int koff = (lane >> 4) << 3; // 0 or 8
  v16h b;
#pragma unroll
  for (int i = 0; i < 16; ++i) {
    b[i] = lds[(kbase + koff + ((i >> 3) << 4) + (i & 7)) * ldn + n];
  }
  return b;
}
// C/D 16x16 f32: lane -> N=lane&15; VGPR i -> M = i + 8*(lane>=16)

// monotone float <-> uint key for atomicMax-based segment max
__device__ __forceinline__ unsigned fkey(float f) {
  unsigned b = __float_as_uint(f);
  return (b & 0x80000000u) ? ~b : (b | 0x80000000u);
}
__device__ __forceinline__ float finv(unsigned k) {
  return (k & 0x80000000u) ? __uint_as_float(k & 0x7fffffffu) : __uint_as_float(~k);
}

// ---------------- sum of edge_attr over edges (for self-loop fill) ----------------
__global__ void k_sumea(const float* __restrict__ ea, float* __restrict__ sumea) {
  __shared__ float s[EDIM];
  int t = threadIdx.x;
  if (t < EDIM) s[t] = 0.f;
  __syncthreads();
  for (int r = blockIdx.x * blockDim.x + t; r < NE; r += gridDim.x * blockDim.x) {
#pragma unroll
    for (int k = 0; k < EDIM; ++k) atomicAdd(&s[k], ea[(size_t)r * EDIM + k]);
  }
  __syncthreads();
  if (t < EDIM) atomicAdd(&sumea[t], s[t]);
}

// ---------------- per-layer weight pre-conversion to f16 (done once, not per block) ----------------
__global__ void k_cvt_f16(const float* __restrict__ gw, const float* __restrict__ pw,
                          _Float16* __restrict__ gw16, _Float16* __restrict__ pw16) {
  int i = blockIdx.x * 256 + threadIdx.x;
  if (i < EMB * HE) {
    gw16[i] = (_Float16)gw[i]; // gat_W  [64,256]
    pw16[i] = (_Float16)pw[i]; // proj_W [256,64]
  }
}

// ---------------- h_t = x_task @ W_task + b_task  (WMMA, K padded 16->32) ----------------
__global__ void __launch_bounds__(128) k_embed_task(const float* __restrict__ xt,
                                                    const float* __restrict__ Wt,
                                                    const float* __restrict__ bt,
                                                    float* __restrict__ ht) {
  __shared__ _Float16 W[32 * 64];
  __shared__ _Float16 A[16 * 32];
  __shared__ float bb[64];
  int tid = threadIdx.x;
  int row0 = blockIdx.x * 16;
  for (int i = tid; i < 32 * 64; i += 128) {
    int k = i >> 6, n = i & 63;
    W[i] = (k < TDIM) ? (_Float16)Wt[k * 64 + n] : (_Float16)0.f;
  }
  for (int i = tid; i < 16 * 32; i += 128) {
    int r = i >> 5, k = i & 31;
    A[i] = (k < TDIM) ? (_Float16)xt[(size_t)(row0 + r) * TDIM + k] : (_Float16)0.f;
  }
  for (int i = tid; i < 64; i += 128) bb[i] = bt[i];
  __syncthreads();
  int lane = tid & 31, wave = tid >> 5; // 4 waves, one 16x16 N-tile each
  v8f acc = {};
  v16h a = frag_a(A, lane, 32, 0);
  v16h b = frag_b(W, lane, 64, 0, wave * 16);
  acc = __builtin_amdgcn_wmma_f32_16x16x32_f16(false, a, false, b, (short)0, acc, false, false);
  int n = lane & 15, rb = (lane >> 4) << 3;
#pragma unroll
  for (int i = 0; i < 8; ++i) {
    int col = wave * 16 + n;
    ht[(size_t)(row0 + rb + i) * EMB + col] = acc[i] + bb[col];
  }
}

// ---------------- h_p = x_proc @ W_proc + b_proc  (+ proc mean-pool accumulate) ----------------
__global__ void k_embed_proc(const float* __restrict__ xp, const float* __restrict__ Wp,
                             const float* __restrict__ bp, const int* __restrict__ pbatch,
                             float* __restrict__ hp, float* __restrict__ pp,
                             float* __restrict__ pcnt) {
  int r = threadIdx.x; // 256 threads, one row each
  int g = pbatch[r];
  atomicAdd(&pcnt[g], 1.f);
  float x[EDIM];
#pragma unroll
  for (int k = 0; k < EDIM; ++k) x[k] = xp[r * EDIM + k];
  for (int c = 0; c < EMB; ++c) {
    float s = bp[c];
#pragma unroll
    for (int k = 0; k < EDIM; ++k) s += x[k] * Wp[k * EMB + c];
    hp[r * EMB + c] = s;
    atomicAdd(&pp[g * EMB + c], s);
  }
}

// ---------------- fold attention vectors into weights (per layer, 1 block) ----------------
__global__ void k_attw(const float* __restrict__ gatW, const float* __restrict__ gatWe,
                       const float* __restrict__ as_, const float* __restrict__ ad_,
                       const float* __restrict__ ae_, const float* __restrict__ sumea,
                       float* __restrict__ wsrc, float* __restrict__ wdst,
                       float* __restrict__ wedge, float* __restrict__ aloop) {
  __shared__ float we[EDIM * HEADS];
  int t = threadIdx.x; // 256 = 64 k * 4 h
  {
    int k = t >> 2, h = t & 3;
    float s1 = 0.f, s2 = 0.f;
    for (int d = 0; d < EMB; ++d) {
      float w = gatW[k * HE + h * EMB + d];
      s1 += w * as_[h * EMB + d];
      s2 += w * ad_[h * EMB + d];
    }
    wsrc[t] = s1;
    wdst[t] = s2;
  }
  if (t < EDIM * HEADS) {
    int k = t >> 2, h = t & 3;
    float s = 0.f;
    for (int d = 0; d < EMB; ++d) s += gatWe[k * HE + h * EMB + d] * ae_[h * EMB + d];
    wedge[t] = s;
    we[t] = s;
  }
  __syncthreads();
  if (t < HEADS) {
    float s = 0.f;
    for (int k = 0; k < EDIM; ++k) s += (sumea[k] * (1.f / (float)NE)) * we[k * HEADS + t];
    aloop[t] = s;
  }
}

// ---------------- xl = h_t @ gat_W (WMMA, async-LDS weights) ; a_src/a_dst ----------------
__global__ void __launch_bounds__(256) k_xl(const float* __restrict__ ht,
                                            const _Float16* __restrict__ gw16,
                                            const float* __restrict__ wsrc,
                                            const float* __restrict__ wdst,
                                            float* __restrict__ xl,
                                            float* __restrict__ asrc,
                                            float* __restrict__ adst) {
  __shared__ __align__(16) _Float16 Wl[EMB * HE];  // 32 KB
  __shared__ _Float16 At[16 * EMB];                // 2 KB
  __shared__ float ws_[EMB * HEADS], wd_[EMB * HEADS];
  int tid = threadIdx.x;
  int row0 = blockIdx.x * 16;
  // async-copy f16 weights straight into LDS (16B per lane per op; 2048 ops total)
  for (int i = tid; i < (EMB * HE) / 8; i += 256)
    async_b128(Wl + i * 8, gw16 + i * 8);
  for (int i = tid; i < 16 * EMB; i += 256) At[i] = (_Float16)ht[(size_t)row0 * EMB + i];
  ws_[tid] = wsrc[tid];
  wd_[tid] = wdst[tid];
  async_wait0();
  __syncthreads();
  int lane = tid & 31, wave = tid >> 5; // 8 waves; 16 N-tiles -> 2 per wave
#pragma unroll
  for (int tile = 0; tile < 2; ++tile) {
    int ntile = wave * 2 + tile;
    v8f acc = {};
#pragma unroll
    for (int kk = 0; kk < 2; ++kk) { // K = 64 = 2 x 32
      v16h a = frag_a(At, lane, EMB, kk * 32);
      v16h b = frag_b(Wl, lane, HE, kk * 32, ntile * 16);
      acc = __builtin_amdgcn_wmma_f32_16x16x32_f16(false, a, false, b, (short)0, acc, false, false);
    }
    int n = lane & 15, rb = (lane >> 4) << 3;
#pragma unroll
    for (int i = 0; i < 8; ++i)
      xl[(size_t)(row0 + rb + i) * HE + ntile * 16 + n] = acc[i];
  }
  // attention logits for these 16 rows
  if (tid < 128) {
    int r = (tid & 63) >> 2, h = tid & 3;
    const float* hrow = ht + (size_t)(row0 + r) * EMB;
    const float* wv = (tid < 64) ? ws_ : wd_;
    float s = 0.f;
    for (int k = 0; k < EMB; ++k) s += hrow[k] * wv[k * HEADS + h];
    if (tid < 64) asrc[(row0 + r) * HEADS + h] = s;
    else          adst[(row0 + r) * HEADS + h] = s;
  }
}

// ---------------- alpha = leaky_relu(asrc[s]+adst[d]+aedg) ; segment max ----------------
__global__ void k_alpha(const int* __restrict__ ei, const float* __restrict__ ea,
                        const float* __restrict__ wedge, const float* __restrict__ aloop,
                        const float* __restrict__ asrc, const float* __restrict__ adst,
                        float* __restrict__ alpha, unsigned* __restrict__ mkey) {
  __shared__ float we[EDIM * HEADS];
  __shared__ float al[HEADS];
  int t = threadIdx.x;
  if (t < EDIM * HEADS) we[t] = wedge[t];
  if (t < HEADS) al[t] = aloop[t];
  __syncthreads();
  int e = blockIdx.x * 256 + t;
  if (e >= ETOT) return;
  int s, d;
  float ag[HEADS];
  if (e < NE) {
    s = ei[e];
    d = ei[NE + e];
    float a[EDIM];
#pragma unroll
    for (int k = 0; k < EDIM; ++k) a[k] = ea[(size_t)e * EDIM + k];
#pragma unroll
    for (int h = 0; h < HEADS; ++h) {
      float x = 0.f;
#pragma unroll
      for (int k = 0; k < EDIM; ++k) x += a[k] * we[k * HEADS + h];
      ag[h] = x;
    }
  } else {
    s = d = e - NE;
#pragma unroll
    for (int h = 0; h < HEADS; ++h) ag[h] = al[h];
  }
#pragma unroll
  for (int h = 0; h < HEADS; ++h) {
    float v = asrc[s * HEADS + h] + adst[d * HEADS + h] + ag[h];
    v = v > 0.f ? v : 0.2f * v; // leaky relu 0.2
    alpha[(size_t)e * HEADS + h] = v;
    atomicMax(&mkey[d * HEADS + h], fkey(v));
  }
}

// ---------------- ex = exp(alpha - max[dst]) ; segment sum denom ----------------
__global__ void k_ex(const int* __restrict__ ei, float* __restrict__ alpha,
                     const unsigned* __restrict__ mkey, float* __restrict__ denom) {
  int e = blockIdx.x * 256 + threadIdx.x;
  if (e >= ETOT) return;
  int d = (e < NE) ? ei[NE + e] : (e - NE);
#pragma unroll
  for (int h = 0; h < HEADS; ++h) {
    float m = finv(mkey[d * HEADS + h]);
    float ex = expf(alpha[(size_t)e * HEADS + h] - m);
    alpha[(size_t)e * HEADS + h] = ex;
    atomicAdd(&denom[d * HEADS + h], ex);
  }
}

// ---------------- agg[dst] += coeff * xl[src]  (one wave per edge, 8 ch/lane) ----------------
__global__ void __launch_bounds__(256) k_agg(const int* __restrict__ ei,
                                             const float* __restrict__ exb,
                                             const float* __restrict__ denom,
                                             const float* __restrict__ xl,
                                             float* __restrict__ agg) {
  int lane = threadIdx.x & 31;
  int e = blockIdx.x * 8 + (threadIdx.x >> 5);
  if (e >= ETOT) return;
  int s, d;
  if (e < NE) { s = ei[e]; d = ei[NE + e]; }
  else        { s = d = e - NE; }
  int h = lane >> 3; // channels lane*8..lane*8+7 share one head
  float coeff = exb[(size_t)e * HEADS + h] / (denom[d * HEADS + h] + 1e-16f);
  int c0 = lane * 8;
  const float4* xs = (const float4*)(xl + (size_t)s * HE + c0);
  float4 x0 = xs[0], x1 = xs[1];
  float* ag = agg + (size_t)d * HE + c0;
  atomicAdd(ag + 0, coeff * x0.x); atomicAdd(ag + 1, coeff * x0.y);
  atomicAdd(ag + 2, coeff * x0.z); atomicAdd(ag + 3, coeff * x0.w);
  atomicAdd(ag + 4, coeff * x1.x); atomicAdd(ag + 5, coeff * x1.y);
  atomicAdd(ag + 6, coeff * x1.z); atomicAdd(ag + 7, coeff * x1.w);
}

// ---------------- fused: elu((agg+gat_b)@proj_W + proj_b) + residual + LayerNorm ----------------
__global__ void __launch_bounds__(128) k_proj_ln(const float* __restrict__ agg,
                                                 const float* __restrict__ gatb,
                                                 const _Float16* __restrict__ pw16,
                                                 const float* __restrict__ projb,
                                                 const float* __restrict__ lng,
                                                 const float* __restrict__ lnb,
                                                 float* __restrict__ ht) {
  __shared__ __align__(16) _Float16 PW[HE * EMB];  // 32 KB
  __shared__ _Float16 At[16 * HE];                 // 8 KB
  __shared__ float hbuf[16 * EMB];
  __shared__ float gb[HE], pb[EMB], lg[EMB], lb[EMB];
  __shared__ float mu[16], rstd[16];
  int tid = threadIdx.x;
  int row0 = blockIdx.x * 16;
  // async-copy pre-converted f16 proj weights straight into LDS
  for (int i = tid; i < (HE * EMB) / 8; i += 128)
    async_b128(PW + i * 8, pw16 + i * 8);
  for (int i = tid; i < HE; i += 128) gb[i] = gatb[i];
  for (int i = tid; i < EMB; i += 128) { pb[i] = projb[i]; lg[i] = lng[i]; lb[i] = lnb[i]; }
  async_wait0();
  __syncthreads();
  for (int i = tid; i < 16 * HE; i += 128) {
    int c = i & 255;
    At[i] = (_Float16)(agg[(size_t)row0 * HE + i] + gb[c]);
  }
  __syncthreads();
  int lane = tid & 31, wave = tid >> 5; // 4 waves, one 16x16 N-tile each
  v8f acc = {};
#pragma unroll
  for (int kk = 0; kk < 8; ++kk) { // K = 256 = 8 x 32
    v16h a = frag_a(At, lane, HE, kk * 32);
    v16h b = frag_b(PW, lane, EMB, kk * 32, wave * 16);
    acc = __builtin_amdgcn_wmma_f32_16x16x32_f16(false, a, false, b, (short)0, acc, false, false);
  }
  int n = lane & 15, rb = (lane >> 4) << 3;
#pragma unroll
  for (int i = 0; i < 8; ++i) {
    int r = rb + i, col = wave * 16 + n;
    float v = acc[i] + pb[col];
    float upd = v > 0.f ? v : expm1f(v); // elu, alpha=1
    hbuf[r * EMB + col] = ht[(size_t)(row0 + r) * EMB + col] + upd;
  }
  __syncthreads();
  if (tid < 16) {
    float s = 0.f;
    for (int c = 0; c < EMB; ++c) s += hbuf[tid * EMB + c];
    float m = s * (1.f / EMB);
    float v = 0.f;
    for (int c = 0; c < EMB; ++c) { float dd = hbuf[tid * EMB + c] - m; v += dd * dd; }
    mu[tid] = m;
    rstd[tid] = rsqrtf(v * (1.f / EMB) + 1e-5f);
  }
  __syncthreads();
  for (int i = tid; i < 16 * EMB; i += 128) {
    int r = i >> 6, c = i & 63;
    ht[(size_t)(row0 + r) * EMB + c] = (hbuf[i] - mu[r]) * rstd[r] * lg[c] + lb[c];
  }
}

// ---------------- task head MLP + task mean-pool accumulate ----------------
__global__ void __launch_bounds__(256) k_task_head(const float* __restrict__ ht,
                                                   const int* __restrict__ tbatch,
                                                   const float* __restrict__ W1,
                                                   const float* __restrict__ b1,
                                                   const float* __restrict__ W2,
                                                   const float* __restrict__ b2,
                                                   float* __restrict__ logits,
                                                   float* __restrict__ tp,
                                                   float* __restrict__ tcnt) {
  __shared__ float w1[EMB * 32];
  __shared__ float bb1[32], w2[32];
  int t = threadIdx.x;
  for (int i = t; i < EMB * 32; i += 256) w1[i] = W1[i];
  if (t < 32) { bb1[t] = b1[t]; w2[t] = W2[t]; }
  __syncthreads();
  int row = blockIdx.x * 256 + t;
  if (row >= NT) return;
  float h[EMB];
#pragma unroll
  for (int k = 0; k < EMB; ++k) h[k] = ht[(size_t)row * EMB + k];
  float acc = b2[0];
  for (int j = 0; j < 32; ++j) {
    float s = bb1[j];
#pragma unroll
    for (int k = 0; k < EMB; ++k) s += h[k] * w1[k * 32 + j];
    acc += (s > 0.f ? s : 0.f) * w2[j];
  }
  logits[row] = acc;
  int g = tbatch[row];
  atomicAdd(&tcnt[g], 1.f);
#pragma unroll
  for (int k = 0; k < EMB; ++k) atomicAdd(&tp[g * EMB + k], h[k]);
}

// ---------------- value head over pooled graph embeddings (1 block) ----------------
__global__ void __launch_bounds__(256) k_value(const float* __restrict__ tp, const float* __restrict__ tcnt,
                                               const float* __restrict__ pp, const float* __restrict__ pcnt,
                                               const float* __restrict__ vW1, const float* __restrict__ vb1,
                                               const float* __restrict__ vW2, const float* __restrict__ vb2,
                                               float* __restrict__ value) {
  __shared__ float ge[NG * 128];
  __shared__ float hid[NG * EMB];
  int t = threadIdx.x;
  for (int i = t; i < NG * EMB; i += 256) {
    int g = i >> 6, c = i & 63;
    ge[g * 128 + c]      = tp[i] / fmaxf(tcnt[g], 1.f);
    ge[g * 128 + 64 + c] = pp[i] / fmaxf(pcnt[g], 1.f);
  }
  __syncthreads();
  for (int i = t; i < NG * EMB; i += 256) {
    int g = i >> 6, j = i & 63;
    float s = vb1[j];
    for (int k = 0; k < 128; ++k) s += ge[g * 128 + k] * vW1[k * EMB + j];
    hid[i] = s > 0.f ? s : 0.f;
  }
  __syncthreads();
  if (t < NG) {
    float s = vb2[0];
    for (int j = 0; j < EMB; ++j) s += hid[t * EMB + j] * vW2[j];
    value[t] = s;
  }
}

// ---------------- workspace layout (float units) ----------------
static const size_t WS_XL    = 0;                                  // NT*HE
static const size_t WS_AGG   = WS_XL + (size_t)NT * HE;            // NT*HE
static const size_t WS_ALPHA = WS_AGG + (size_t)NT * HE;           // ETOT*HEADS
static const size_t WS_ASRC  = WS_ALPHA + (size_t)ETOT * HEADS;    // NT*HEADS
static const size_t WS_ADST  = WS_ASRC + (size_t)NT * HEADS;       // NT*HEADS
static const size_t WS_MKEY  = WS_ADST + (size_t)NT * HEADS;       // NT*HEADS (uint)
static const size_t WS_DEN   = WS_MKEY + (size_t)NT * HEADS;       // NT*HEADS
static const size_t WS_SUMEA = WS_DEN + (size_t)NT * HEADS;        // EDIM
static const size_t WS_WSRC  = WS_SUMEA + EDIM;                    // EMB*HEADS
static const size_t WS_WDST  = WS_WSRC + EMB * HEADS;
static const size_t WS_WEDGE = WS_WDST + EMB * HEADS;              // EDIM*HEADS
static const size_t WS_ALOOP = WS_WEDGE + EDIM * HEADS;            // HEADS
static const size_t WS_TP    = WS_ALOOP + HEADS;                   // NG*EMB
static const size_t WS_PP    = WS_TP + NG * EMB;                   // NG*EMB
static const size_t WS_TCNT  = WS_PP + NG * EMB;                   // NG
static const size_t WS_PCNT  = WS_TCNT + NG;                       // NG
static const size_t WS_GW16  = WS_PCNT + NG;                       // EMB*HE halves = 8192 floats
static const size_t WS_PW16  = WS_GW16 + (EMB * HE) / 2;           // HE*EMB halves = 8192 floats
static const size_t WS_TOTAL = WS_PW16 + (HE * EMB) / 2;

extern "C" void kernel_launch(void* const* d_in, const int* in_sizes, int n_in,
                              void* d_out, int out_size, void* d_ws, size_t ws_size,
                              hipStream_t stream) {
  (void)in_sizes; (void)n_in; (void)out_size; (void)ws_size;
  const float* x_task     = (const float*)d_in[0];
  const float* x_proc     = (const float*)d_in[1];
  const int*   edge_index = (const int*)d_in[2];
  const float* edge_attr  = (const float*)d_in[3];
  const int*   task_batch = (const int*)d_in[4];
  const int*   proc_batch = (const int*)d_in[5];
  const float* W_task = (const float*)d_in[6];
  const float* b_task = (const float*)d_in[7];
  const float* W_proc = (const float*)d_in[8];
  const float* b_proc = (const float*)d_in[9];
  const float* gat_W    = (const float*)d_in[10];
  const float* gat_We   = (const float*)d_in[11];
  const float* att_src  = (const float*)d_in[12];
  const float* att_dst  = (const float*)d_in[13];
  const float* att_edge = (const float*)d_in[14];
  const float* gat_b    = (const float*)d_in[15];
  const float* proj_W   = (const float*)d_in[16];
  const float* proj_b   = (const float*)d_in[17];
  const float* ln_g     = (const float*)d_in[18];
  const float* ln_b     = (const float*)d_in[19];
  const float* pt_W1 = (const float*)d_in[20];
  const float* pt_b1 = (const float*)d_in[21];
  const float* pt_W2 = (const float*)d_in[22];
  const float* pt_b2 = (const float*)d_in[23];
  const float* v_W1 = (const float*)d_in[24];
  const float* v_b1 = (const float*)d_in[25];
  const float* v_W2 = (const float*)d_in[26];
  const float* v_b2 = (const float*)d_in[27];

  float* out    = (float*)d_out;
  float* logits = out;                                   // [NT]
  float* value  = out + NT;                              // [NG]
  float* ht     = out + NT + NG;                         // [NT,EMB] (working + final)
  float* hp     = out + NT + NG + (size_t)NT * EMB;      // [NP,EMB]

  float* ws = (float*)d_ws;

  // zero the small accumulators (sumea .. pcnt, contiguous tail region)
  hipMemsetAsync(ws + WS_SUMEA, 0, (WS_GW16 - WS_SUMEA) * sizeof(float), stream);

  k_sumea<<<1024, 256, 0, stream>>>(edge_attr, ws + WS_SUMEA);
  k_embed_task<<<NT / 16, 128, 0, stream>>>(x_task, W_task, b_task, ht);
  k_embed_proc<<<1, 256, 0, stream>>>(x_proc, W_proc, b_proc, proc_batch,
                                      hp, ws + WS_PP, ws + WS_PCNT);

  const int EB = (ETOT + 255) / 256;
  for (int l = 0; l < 3; ++l) {
    // zero max-keys + denoms (contiguous) and the aggregation buffer
    hipMemsetAsync(ws + WS_MKEY, 0, (size_t)NT * HEADS * 2 * sizeof(float), stream);
    hipMemsetAsync(ws + WS_AGG, 0, (size_t)NT * HE * sizeof(float), stream);

    k_cvt_f16<<<(EMB * HE + 255) / 256, 256, 0, stream>>>(
        gat_W + (size_t)l * EMB * HE, proj_W + (size_t)l * HE * EMB,
        (_Float16*)(ws + WS_GW16), (_Float16*)(ws + WS_PW16));
    k_attw<<<1, 256, 0, stream>>>(gat_W + (size_t)l * EMB * HE,
                                  gat_We + (size_t)l * EDIM * HE,
                                  att_src + (size_t)l * HEADS * EMB,
                                  att_dst + (size_t)l * HEADS * EMB,
                                  att_edge + (size_t)l * HEADS * EMB,
                                  ws + WS_SUMEA, ws + WS_WSRC, ws + WS_WDST,
                                  ws + WS_WEDGE, ws + WS_ALOOP);
    k_xl<<<NT / 16, 256, 0, stream>>>(ht, (const _Float16*)(ws + WS_GW16),
                                      ws + WS_WSRC, ws + WS_WDST,
                                      ws + WS_XL, ws + WS_ASRC, ws + WS_ADST);
    k_alpha<<<EB, 256, 0, stream>>>(edge_index, edge_attr, ws + WS_WEDGE, ws + WS_ALOOP,
                                    ws + WS_ASRC, ws + WS_ADST,
                                    ws + WS_ALPHA, (unsigned*)(ws + WS_MKEY));
    k_ex<<<EB, 256, 0, stream>>>(edge_index, ws + WS_ALPHA,
                                 (const unsigned*)(ws + WS_MKEY), ws + WS_DEN);
    k_agg<<<(ETOT + 7) / 8, 256, 0, stream>>>(edge_index, ws + WS_ALPHA, ws + WS_DEN,
                                              ws + WS_XL, ws + WS_AGG);
    k_proj_ln<<<NT / 16, 128, 0, stream>>>(ws + WS_AGG, gat_b + (size_t)l * HE,
                                           (const _Float16*)(ws + WS_PW16),
                                           proj_b + (size_t)l * EMB,
                                           ln_g + (size_t)l * EMB, ln_b + (size_t)l * EMB, ht);
  }

  k_task_head<<<(NT + 255) / 256, 256, 0, stream>>>(ht, task_batch, pt_W1, pt_b1, pt_W2, pt_b2,
                                                    logits, ws + WS_TP, ws + WS_TCNT);
  k_value<<<1, 256, 0, stream>>>(ws + WS_TP, ws + WS_TCNT, ws + WS_PP, ws + WS_PCNT,
                                 v_W1, v_b1, v_W2, v_b2, value);
}
